// snnlayer_47983374631234
// MI455X (gfx1250) — matrix operand order repaired and checked by reading, not compiled, not run on hardware
//
#include <hip/hip_runtime.h>

#define N_EVENTS  2048
#define TS_SIZE   1024
#define N_NEURONS 128
#define NTHREADS  256   // 8 waves (wave32): wave w owns neuron rows 16w..16w+15

typedef __bf16 bf16_t;
typedef __attribute__((ext_vector_type(16))) bf16_t v16bf;
typedef __attribute__((ext_vector_type(8)))  bf16_t v8bf;
typedef __attribute__((ext_vector_type(8)))  float  v8f;

// ---- dynamic LDS layout (bytes) ----
#define OFF_WB    0         // bf16 W mirror: 128*1024*2 = 262144
#define OFF_TS    262144    // normalized ts, f32: 4096
#define OFF_TSB   266240    // normalized ts, bf16: 2048
#define OFF_DOT   268288    // raw dots -> normalized beta: 512
#define OFF_LG    268800    // logits: 512
#define OFF_RED   269312    // 256-wide reduction scratch: 1024
#define OFF_MASK  270336    // winner mask (int): 512
#define OFF_N2    270848    // row norm^2: 512
#define OFF_H     271360    // cumhisto h: 512
#define OFF_SCAL  271872    // [0]=h_sum, [1]=norm2 accumulator
#define OFF_INT   271904    // [0]=first winner index
#define OFF_RAW   271936    // raw ts double buffer: 2 * 4096 (async-to-LDS staging)
#define SMEM_BYTES 280128   // < 320 KB WGP LDS

// Async DMA of one 4 KB ts row into LDS: one b128 per lane, 256 lanes total.
__device__ __forceinline__ void async_load_ts(const float* gbase,
                                              unsigned lds_base, int tid) {
  const unsigned dst  = lds_base + (unsigned)(tid << 4);   // LDS byte offset
  const unsigned voff = (unsigned)(tid << 4);              // per-lane byte offset
  asm volatile("global_load_async_to_lds_b128 %0, %1, %2"
               :: "v"(dst), "v"(voff),
                  "s"((unsigned long long)gbase)
               : "memory");
}

__global__ __launch_bounds__(NTHREADS, 1)
void snn_scan_kernel(const float* __restrict__ all_ts,
                     const float* __restrict__ W_in,
                     const float* __restrict__ h_in,
                     float* __restrict__ out_nstar,   // [2048]
                     float* __restrict__ out_W,       // [128*1024] fp32 master (in-place)
                     float* __restrict__ out_h)       // [128]
{
  extern __shared__ char smem[];
  bf16_t* sWb  = (bf16_t*)(smem + OFF_WB);
  float*  sTs  = (float*) (smem + OFF_TS);
  bf16_t* sTsb = (bf16_t*)(smem + OFF_TSB);
  float*  sDot = (float*) (smem + OFF_DOT);
  float*  sLg  = (float*) (smem + OFF_LG);
  float*  sRed = (float*) (smem + OFF_RED);
  int*    sMask= (int*)   (smem + OFF_MASK);
  float*  sN2  = (float*) (smem + OFF_N2);
  float*  sH   = (float*) (smem + OFF_H);
  float*  sScal= (float*) (smem + OFF_SCAL);
  int*    sInt = (int*)   (smem + OFF_INT);
  float*  sRaw = (float*) (smem + OFF_RAW);
  // Wave-relative LDS byte offset of the raw staging area (low 32 bits of the
  // generic pointer are the LDS address per gfx1250 flat addressing).
  const unsigned rawOff = (unsigned)(size_t)sRaw;

  const int tid  = threadIdx.x;
  const int lane = tid & 31;
  const int wave = tid >> 5;

  // ---------------- init: fp32 master -> out_W, bf16 mirror -> LDS, row norms, h ----
  // Kick off the async DMA of event 0's ts row immediately.
  async_load_ts(all_ts, rawOff, tid);

  if (tid < N_NEURONS) { sN2[tid] = 0.0f; sH[tid] = h_in[tid]; }
  __syncthreads();
  for (int i = tid; i < (N_NEURONS * TS_SIZE) / 4; i += NTHREADS) {
    const float4 v = ((const float4*)W_in)[i];
    ((float4*)out_W)[i] = v;
    const int base = i << 2;
    sWb[base+0] = (bf16_t)v.x; sWb[base+1] = (bf16_t)v.y;
    sWb[base+2] = (bf16_t)v.z; sWb[base+3] = (bf16_t)v.w;
    atomicAdd(&sN2[base >> 10], v.x*v.x + v.y*v.y + v.z*v.z + v.w*v.w);
  }
  sRed[tid] = (tid < N_NEURONS) ? sH[tid] : 0.0f;
  __syncthreads();
  for (int s = NTHREADS/2; s > 0; s >>= 1) {
    if (tid < s) sRed[tid] += sRed[tid + s];
    __syncthreads();
  }
  if (tid == 0) sScal[0] = sRed[0];   // h_sum
  __syncthreads();

  // ---------------- sequential event scan ----------------
  for (int e = 0; e < N_EVENTS; ++e) {
    const int buf = e & 1;
    // Prefetch next event's ts into the other buffer; wait until the current
    // buffer's DMA has landed (async loads complete in order -> <=1 outstanding).
    if (e + 1 < N_EVENTS) {
      async_load_ts(all_ts + (size_t)(e + 1) * TS_SIZE,
                    rawOff + (unsigned)(((e + 1) & 1) * (TS_SIZE * 4)), tid);
      asm volatile("s_wait_asynccnt 0x1" ::: "memory");
    } else {
      asm volatile("s_wait_asynccnt 0x0" ::: "memory");
    }
    __syncthreads();   // all waves' async portions visible in LDS

    // 1. normalize event time surface from the LDS staging buffer
    float4 tv = ((const float4*)(sRaw + buf * TS_SIZE))[tid];
    sRed[tid] = tv.x*tv.x + tv.y*tv.y + tv.z*tv.z + tv.w*tv.w;
    __syncthreads();
    for (int s = NTHREADS/2; s > 0; s >>= 1) {
      if (tid < s) sRed[tid] += sRed[tid + s];
      __syncthreads();
    }
    const float inv = 1.0f / sqrtf(sRed[0]);
    tv.x *= inv; tv.y *= inv; tv.z *= inv; tv.w *= inv;
    ((float4*)sTs)[tid] = tv;
    const int kb4 = tid << 2;
    sTsb[kb4+0] = (bf16_t)tv.x; sTsb[kb4+1] = (bf16_t)tv.y;
    sTsb[kb4+2] = (bf16_t)tv.z; sTsb[kb4+3] = (bf16_t)tv.w;
    __syncthreads();

    // 2. dot[n] = W[n,:] . ts via v_wmma_f32_16x16x32_bf16 (B = ts replicated over 16 cols)
    const int rowb = wave << 4;
    const bf16_t* wrow = sWb + (size_t)(rowb + (lane & 15)) * TS_SIZE;
    const int hi = lane >> 4;
    v8f acc = {};
#pragma unroll 4
    for (int t = 0; t < TS_SIZE / 32; ++t) {
      // A tile 16x32: lane holds K in {k0..k0+15}, k0 = 32t + 16*hi
      const int kA = t*32 + (hi << 4);
      v8bf a0 = *(const v8bf*)(wrow + kA);
      v8bf a1 = *(const v8bf*)(wrow + kA + 8);
      v16bf A = __builtin_shufflevector(a0, a1,
                    0,1,2,3,4,5,6,7,8,9,10,11,12,13,14,15);
      // B tile 32x16 (all cols equal): lane group hi picks K-half, 32 contiguous bytes
      const int kB = t*32 + (hi << 4);
      v16bf B = *(const v16bf*)(sTsb + kB);
      acc = __builtin_amdgcn_wmma_f32_16x16x32_bf16(
                false, A, false, B, (short)0, acc, false, false);
    }
    // C layout: VGPR r -> M=r (lanes 0-15), M=r+8 (lanes 16-31); all N identical
    if (lane == 0) {
#pragma unroll
      for (int r = 0; r < 8; ++r) sDot[rowb + r] = acc[r];
    }
    if (lane == 16) {
#pragma unroll
      for (int r = 0; r < 8; ++r) sDot[rowb + 8 + r] = acc[r];
    }
    if (tid == 0) sInt[0] = 1 << 30;
    __syncthreads();

    // 3. beta, homeostatic gain, softmax over 128 neurons, winner mask
    float logit = -1e30f;
    if (tid < N_NEURONS) {
      const float beta = sDot[tid] / sqrtf(sN2[tid]);
      sDot[tid] = beta;                       // keep normalized beta for the update
      const float gain =
          __expf(0.25f * (1.0f - (float)N_NEURONS * sH[tid] / sScal[0]));
      logit = gain * beta;
      sLg[tid] = logit;
    }
    sRed[tid] = logit;
    __syncthreads();
    for (int s = NTHREADS/2; s > 0; s >>= 1) {          // max
      if (tid < s) sRed[tid] = fmaxf(sRed[tid], sRed[tid + s]);
      __syncthreads();
    }
    const float mx = sRed[0];
    float ex = 0.0f;
    if (tid < N_NEURONS) { ex = __expf(sLg[tid] - mx); }
    sRed[tid] = ex;
    __syncthreads();
    for (int s = NTHREADS/2; s > 0; s >>= 1) {          // sum
      if (tid < s) sRed[tid] += sRed[tid + s];
      __syncthreads();
    }
    const float esum = sRed[0];
    if (tid < N_NEURONS) {
      const int win = ((ex / esum) > (5.0f / (float)N_NEURONS)) ? 1 : 0;
      sMask[tid] = win;
      if (win) atomicMin(&sInt[0], tid);      // first winner (argmax of bool mask)
    }
    __syncthreads();
    if (tid == 0) {
      const int f = sInt[0];
      out_nstar[e] = (f < N_NEURONS) ? (float)f : -1.0f;
    }

    // 4. scatter-update winner rows: W += alpha*beta*(ts - W); refresh norm2, mirror, h
    for (int n = 0; n < N_NEURONS; ++n) {
      if (sMask[n]) {                         // uniform branch (LDS value after barrier)
        if (tid == 0) sScal[1] = 0.0f;
        __syncthreads();
        const float hn   = sH[n];
        const float coef = (0.01f / (1.0f + hn * (1.0f / 20000.0f))) * sDot[n];
        float4 wv        = ((float4*)(out_W + (size_t)n * TS_SIZE))[tid];
        const float4 tsv = ((const float4*)sTs)[tid];
        wv.x += coef * (tsv.x - wv.x);
        wv.y += coef * (tsv.y - wv.y);
        wv.z += coef * (tsv.z - wv.z);
        wv.w += coef * (tsv.w - wv.w);
        ((float4*)(out_W + (size_t)n * TS_SIZE))[tid] = wv;
        const int kb = n * TS_SIZE + (tid << 2);
        sWb[kb+0] = (bf16_t)wv.x; sWb[kb+1] = (bf16_t)wv.y;
        sWb[kb+2] = (bf16_t)wv.z; sWb[kb+3] = (bf16_t)wv.w;
        atomicAdd(&sScal[1], wv.x*wv.x + wv.y*wv.y + wv.z*wv.z + wv.w*wv.w);
        __syncthreads();
        if (tid == 0) {
          sN2[n] = sScal[1];                  // exact fp32 norm of updated row
          sH[n]  = hn + 1.0f;
          sScal[0] += 1.0f;                   // h_sum
        }
        __syncthreads();
      }
    }
    __syncthreads();
  }

  if (tid < N_NEURONS) out_h[tid] = sH[tid];
}

extern "C" void kernel_launch(void* const* d_in, const int* in_sizes, int n_in,
                              void* d_out, int out_size, void* d_ws, size_t ws_size,
                              hipStream_t stream) {
  (void)in_sizes; (void)n_in; (void)out_size; (void)d_ws; (void)ws_size;
  const float* all_ts = (const float*)d_in[0];
  const float* W      = (const float*)d_in[1];
  const float* h0     = (const float*)d_in[2];
  // d_in[3] = clustering_flag (always 1 path) — unused.

  float* out_nstar = (float*)d_out;                          // [2048]
  float* out_W     = out_nstar + N_EVENTS;                   // [128*1024]
  float* out_h     = out_W + (size_t)N_NEURONS * TS_SIZE;    // [128]

  // Allow >default dynamic LDS (280 KB of the 320 KB WGP LDS). Deterministic, idempotent.
  hipFuncSetAttribute((const void*)snn_scan_kernel,
                      hipFuncAttributeMaxDynamicSharedMemorySize, SMEM_BYTES);

  snn_scan_kernel<<<1, NTHREADS, SMEM_BYTES, stream>>>(
      all_ts, W, h0, out_nstar, out_W, out_h);
}
